// NeuralFP_52029233824314
// MI455X (gfx1250) — matrix operand order repaired and checked by reading, compiled, not run on hardware
//
#include <hip/hip_runtime.h>
#include <hip/hip_bf16.h>

typedef __attribute__((ext_vector_type(16))) __bf16 v16bf;
typedef __attribute__((ext_vector_type(8)))  float  v8f;

#define ATOM_F 64
#define FPSZ   2048
#define LGS    (FPSZ + 4)   // padded LDS row stride (floats) -> bank-conflict-free stores

__device__ __forceinline__ v8f wmma_bf16(v16bf a, v16bf b, v8f c) {
  // D = A(16x32 bf16) * B(32x16 bf16) + C(16x16 f32)
  return __builtin_amdgcn_wmma_f32_16x16x32_bf16(false, a, false, b, (short)0, c, false, false);
}

__device__ __forceinline__ float fast_rcp(float x) {
  return __builtin_amdgcn_rcpf(x);               // single v_rcp_f32
}

__device__ __forceinline__ float fast_sigmoid(float x) {
  return fast_rcp(1.f + __expf(-x));
}

// ---------------- utility kernels ----------------

__global__ void k_zero(float* __restrict__ p, int n) {
  int i = blockIdx.x * blockDim.x + threadIdx.x;
  if (i < n) p[i] = 0.f;
}

__global__ void k_copy4(const float4* __restrict__ s, float4* __restrict__ d, long long n4) {
  long long i = (long long)blockIdx.x * blockDim.x + threadIdx.x;
  if (i < n4) d[i] = s[i];
}

// Pack W [OUT, 64] (f32, row-major; out = A @ W^T so B[k][n] = W[n][k]) into
// per-lane WMMA B-fragments: frag[((tile*2+kt)*32 + lane)*16 + j] (bf16).
__global__ void k_pack_w(const float* __restrict__ W, __bf16* __restrict__ frag, int ntiles) {
  int t = blockIdx.x * blockDim.x + threadIdx.x;   // over ntiles*2*32 lane-fragments
  if (t >= ntiles * 64) return;
  int lane = t & 31;
  int kt   = (t >> 5) & 1;
  int tile = t >> 6;
  int n     = tile * 16 + (lane & 15);
  int kbase = kt * 32 + ((lane >> 4) << 3);
  const float* wr = W + (size_t)n * ATOM_F + kbase;
  __bf16* o = frag + (size_t)t * 16;
#pragma unroll
  for (int j = 0; j < 16; ++j) o[j] = (__bf16)wr[j + ((j >> 3) << 3)];
}

// ---------------- edge scatter-add: agg[dst] += h[src] ----------------
__global__ void k_edge(const float* __restrict__ h, const long long* __restrict__ ei,
                       float* __restrict__ agg, long long n_edges) {
  long long tid = (long long)blockIdx.x * blockDim.x + threadIdx.x;
  if (tid >= n_edges * 16) return;
  long long e = tid >> 4;
  int q = (int)(tid & 15);
  long long s = ei[e];
  long long d = ei[n_edges + e];
  const float4 v = *(const float4*)(h + s * ATOM_F + q * 4);
  float* a = agg + d * ATOM_F + q * 4;
  unsafeAtomicAdd(a + 0, v.x);
  unsafeAtomicAdd(a + 1, v.y);
  unsafeAtomicAdd(a + 2, v.z);
  unsafeAtomicAdd(a + 3, v.w);
}

// ---------------- H layer: upd = sigmoid(agg @ Hw^T + b), also emit bf16 A-fragments ----------------
__launch_bounds__(256)
__global__ void k_hlayer(const float* __restrict__ agg, const __bf16* __restrict__ hfrag,
                         const float* __restrict__ bias, float* __restrict__ upd,
                         __bf16* __restrict__ ufrag, long long n_nodes, int n_rt) {
  __shared__ float tile[8][16 * ATOM_F];          // 32 KB
  int lane = threadIdx.x & 31;
  int w    = threadIdx.x >> 5;
  int rt   = blockIdx.x * 8 + w;
  bool active = rt < n_rt;
  int rt_c = active ? rt : 0;

  int h    = lane >> 4;                            // lane half
  int rloc = lane & 15;
  long long row = (long long)rt_c * 16 + rloc;
  if (row >= n_nodes) row = n_nodes - 1;

  // A fragments (16x32 bf16): lane L row = L&15; VGPR v: K = (v>>2)*16 + h*8 + (v&3)*2 (+1)
  v16bf a[2];
#pragma unroll
  for (int kt = 0; kt < 2; ++kt) {
    const float* rp = agg + row * ATOM_F + kt * 32;
#pragma unroll
    for (int v = 0; v < 8; ++v) {
      int k = ((v >> 2) << 4) + (h << 3) + ((v & 3) << 1);
      a[kt][2 * v]     = (__bf16)rp[k];
      a[kt][2 * v + 1] = (__bf16)rp[k + 1];
    }
  }

  v8f acc[4];
#pragma unroll
  for (int t = 0; t < 4; ++t) acc[t] = {};
#pragma unroll
  for (int kt = 0; kt < 2; ++kt) {
#pragma unroll
    for (int t = 0; t < 4; ++t) {
      v16bf b = *(const v16bf*)(hfrag + (size_t)((t * 2 + kt) * 32 + lane) * 16);
      acc[t] = wmma_bf16(a[kt], b, acc[t]);
    }
  }

  // bias + sigmoid; D layout: lane L col = L&15; VGPR v row = v + 8*(L>>4)
  float* my = tile[w];
#pragma unroll
  for (int t = 0; t < 4; ++t) {
    int col = t * 16 + rloc;
    float bv = bias[col];
#pragma unroll
    for (int v = 0; v < 8; ++v) {
      int rl = v + (h << 3);
      float sgm = fast_sigmoid(acc[t][v] + bv);
      my[rl * ATOM_F + col] = sgm;
      long long r = (long long)rt_c * 16 + rl;
      if (active && r < n_nodes) upd[r * ATOM_F + col] = sgm;
    }
  }
  __syncthreads();

  // Repack sigmoid output into A-fragment layout for the FP GEMM
  if (active) {
#pragma unroll
    for (int kt = 0; kt < 2; ++kt) {
      v16bf o;
      const float* rp2 = my + rloc * ATOM_F + kt * 32;
#pragma unroll
      for (int v = 0; v < 8; ++v) {
        int k = ((v >> 2) << 4) + (h << 3) + ((v & 3) << 1);
        o[2 * v]     = (__bf16)rp2[k];
        o[2 * v + 1] = (__bf16)rp2[k + 1];
      }
      *(v16bf*)(ufrag + (size_t)((size_t)(rt * 2 + kt) * 32 + lane) * 16) = o;
    }
  }
}

// ---------------- fused FP layer: softmax(upd @ Ww^T + b) pooled by graph ----------------
__launch_bounds__(256)
__global__ void k_fp(const __bf16* __restrict__ ufrag, const __bf16* __restrict__ wfrag,
                     const float* __restrict__ bias, const long long* __restrict__ batch,
                     float* __restrict__ out, long long n_nodes) {
  extern __shared__ float lg[];                  // 16 * LGS f32 (padded rows)
  __shared__ float red[16 * 16];
  __shared__ float rowmax[16], rowinv[16];
  __shared__ int gid[16];
  __shared__ int gsame;

  int rt   = blockIdx.x;
  int lane = threadIdx.x & 31;
  int w    = threadIdx.x >> 5;
  int rloc = lane & 15;
  int h    = lane >> 4;

  v16bf a0 = *(const v16bf*)(ufrag + (size_t)((size_t)(rt * 2 + 0) * 32 + lane) * 16);
  v16bf a1 = *(const v16bf*)(ufrag + (size_t)((size_t)(rt * 2 + 1) * 32 + lane) * 16);

  for (int i = 0; i < 16; ++i) {
    int ct = w * 16 + i;
    const __bf16* bp0 = wfrag + (size_t)((size_t)(ct * 2 + 0) * 32 + lane) * 16;
    const __bf16* bp1 = wfrag + (size_t)((size_t)(ct * 2 + 1) * 32 + lane) * 16;
    if (i < 15) {   // prefetch this tile's kt=1 and next tile's kt=0 fragments (global_prefetch_b8)
      __builtin_prefetch(bp0 + 32 * 16, 0, 3);
      __builtin_prefetch(bp1 + 32 * 16, 0, 3);
    }
    v16bf b0 = *(const v16bf*)bp0;
    v16bf b1 = *(const v16bf*)bp1;
    v8f acc = {};
    acc = wmma_bf16(a0, b0, acc);
    acc = wmma_bf16(a1, b1, acc);
    int col = ct * 16 + rloc;
    float bv = bias[col];
#pragma unroll
    for (int v = 0; v < 8; ++v) {
      int rl = v + (h << 3);
      lg[rl * LGS + col] = acc[v] + bv;   // padded stride: conflict-free across lane halves
    }
  }
  if (threadIdx.x < 16) {
    long long r = (long long)rt * 16 + threadIdx.x;
    gid[threadIdx.x] = (r < n_nodes) ? (int)batch[r] : -1;
  }
  __syncthreads();

  // row max (float4 LDS traffic: ds_load_b128)
  {
    int row = threadIdx.x >> 4, seg = threadIdx.x & 15;
    const float4* p4 = (const float4*)(lg + row * LGS + seg * 128);
    float m = -1e30f;
    for (int j = 0; j < 32; ++j) {
      float4 v = p4[j];
      m = fmaxf(m, fmaxf(fmaxf(v.x, v.y), fmaxf(v.z, v.w)));
    }
    red[row * 16 + seg] = m;
  }
  __syncthreads();
  if (threadIdx.x < 16) {
    float m = -1e30f;
    for (int j = 0; j < 16; ++j) m = fmaxf(m, red[threadIdx.x * 16 + j]);
    rowmax[threadIdx.x] = m;
  }
  __syncthreads();

  // exp in place + row sum (float4)
  {
    int row = threadIdx.x >> 4, seg = threadIdx.x & 15;
    float4* p4 = (float4*)(lg + row * LGS + seg * 128);
    float m = rowmax[row], s = 0.f;
    for (int j = 0; j < 32; ++j) {
      float4 v = p4[j];
      v.x = __expf(v.x - m); v.y = __expf(v.y - m);
      v.z = __expf(v.z - m); v.w = __expf(v.w - m);
      p4[j] = v;
      s += (v.x + v.y) + (v.z + v.w);
    }
    red[row * 16 + seg] = s;
  }
  __syncthreads();
  if (threadIdx.x < 16) {
    float s = 0.f;
    for (int j = 0; j < 16; ++j) s += red[threadIdx.x * 16 + j];
    rowinv[threadIdx.x] = fast_rcp(s);
    if (threadIdx.x == 0) {
      int g0 = gid[0], same = 1;
      for (int j = 1; j < 16; ++j) if (gid[j] != g0) same = 0;
      gsame = same && (g0 >= 0);
    }
  }
  __syncthreads();

  if (gsame) {
    // single-graph tile: pre-reduce 16 rows in LDS, 1 atomic per column
    int g0 = gid[0];
    for (int c4 = threadIdx.x; c4 < FPSZ / 4; c4 += 256) {
      float4 s = {0.f, 0.f, 0.f, 0.f};
#pragma unroll
      for (int r = 0; r < 16; ++r) {
        const float4 v = *(const float4*)(lg + r * LGS + c4 * 4);
        float iv = rowinv[r];
        s.x += v.x * iv; s.y += v.y * iv; s.z += v.z * iv; s.w += v.w * iv;
      }
      float* o = out + (size_t)g0 * FPSZ + c4 * 4;
      unsafeAtomicAdd(o + 0, s.x);
      unsafeAtomicAdd(o + 1, s.y);
      unsafeAtomicAdd(o + 2, s.z);
      unsafeAtomicAdd(o + 3, s.w);
    }
  } else {
    int row = threadIdx.x >> 4, seg = threadIdx.x & 15;
    int g = gid[row];
    if (g >= 0) {
      float inv = rowinv[row];
      const float4* p4 = (const float4*)(lg + row * LGS + seg * 128);
      float* o = out + (size_t)g * FPSZ + seg * 128;
      for (int j = 0; j < 32; ++j) {
        float4 v = p4[j];
        unsafeAtomicAdd(o + 4 * j + 0, v.x * inv);
        unsafeAtomicAdd(o + 4 * j + 1, v.y * inv);
        unsafeAtomicAdd(o + 4 * j + 2, v.z * inv);
        unsafeAtomicAdd(o + 4 * j + 3, v.w * inv);
      }
    }
  }
}

// ---------------- launch ----------------

extern "C" void kernel_launch(void* const* d_in, const int* in_sizes, int n_in,
                              void* d_out, int out_size, void* d_ws, size_t ws_size,
                              hipStream_t stream) {
  const float* x   = (const float*)d_in[0];
  const float* H1w = (const float*)d_in[1];
  const float* H1b = (const float*)d_in[2];
  const float* W1w = (const float*)d_in[3];
  const float* W1b = (const float*)d_in[4];
  const float* H2w = (const float*)d_in[5];
  const float* H2b = (const float*)d_in[6];
  const float* W2w = (const float*)d_in[7];
  const float* W2b = (const float*)d_in[8];
  const long long* ei    = (const long long*)d_in[9];
  const long long* batch = (const long long*)d_in[10];
  float* out = (float*)d_out;

  long long n_nodes = (long long)in_sizes[0] / ATOM_F;
  long long n_edges = (long long)in_sizes[9] / 2;
  int n_rt = (int)((n_nodes + 15) / 16);
  const int NWT = FPSZ / 16;   // 128 col tiles for the FP layer

  char* ws = (char*)d_ws;
  size_t off = 0;
  auto wsalloc = [&](size_t bytes) -> char* {
    char* p = ws + off; off += (bytes + 255) & ~(size_t)255; return p;
  };
  float*  agg   = (float*)wsalloc((size_t)n_nodes * ATOM_F * 4);
  float*  upd   = (float*)wsalloc((size_t)n_nodes * ATOM_F * 4);
  __bf16* ufrag = (__bf16*)wsalloc((size_t)n_rt * 2 * 32 * 16 * 2);
  __bf16* h1f   = (__bf16*)wsalloc((size_t)4 * 2 * 32 * 16 * 2);
  __bf16* h2f   = (__bf16*)wsalloc((size_t)4 * 2 * 32 * 16 * 2);
  __bf16* w1f   = (__bf16*)wsalloc((size_t)NWT * 2 * 32 * 16 * 2);
  __bf16* w2f   = (__bf16*)wsalloc((size_t)NWT * 2 * 32 * 16 * 2);

  k_zero<<<(out_size + 255) / 256, 256, 0, stream>>>(out, out_size);

  k_pack_w<<<(4 * 64 + 255) / 256, 256, 0, stream>>>(H1w, h1f, 4);
  k_pack_w<<<(4 * 64 + 255) / 256, 256, 0, stream>>>(H2w, h2f, 4);
  k_pack_w<<<(NWT * 64 + 255) / 256, 256, 0, stream>>>(W1w, w1f, NWT);
  k_pack_w<<<(NWT * 64 + 255) / 256, 256, 0, stream>>>(W2w, w2f, NWT);

  long long n4 = n_nodes * ATOM_F / 4;
  long long ethreads = n_edges * 16;
  size_t fp_lds = (size_t)16 * LGS * sizeof(float);   // ~128 KB dynamic LDS (padded)

  // ---- layer 1 ----
  k_copy4<<<(int)((n4 + 255) / 256), 256, 0, stream>>>((const float4*)x, (float4*)agg, n4);
  k_edge<<<(int)((ethreads + 255) / 256), 256, 0, stream>>>(x, ei, agg, n_edges);
  k_hlayer<<<(n_rt + 7) / 8, 256, 0, stream>>>(agg, h1f, H1b, upd, ufrag, n_nodes, n_rt);
  k_fp<<<n_rt, 256, fp_lds, stream>>>(ufrag, w1f, W1b, batch, out, n_nodes);

  // ---- layer 2 (h = upd from layer 1) ----
  k_copy4<<<(int)((n4 + 255) / 256), 256, 0, stream>>>((const float4*)upd, (float4*)agg, n4);
  k_edge<<<(int)((ethreads + 255) / 256), 256, 0, stream>>>(upd, ei, agg, n_edges);
  k_hlayer<<<(n_rt + 7) / 8, 256, 0, stream>>>(agg, h2f, H2b, upd, ufrag, n_nodes, n_rt);
  k_fp<<<n_rt, 256, fp_lds, stream>>>(ufrag, w2f, W2b, batch, out, n_nodes);
}
